// RealTimeWindowPredictor_64957085384888
// MI455X (gfx1250) — compile-verified, matrix operand order checked by reading, not run on hardware
//
#include <hip/hip_runtime.h>
#include <hip/hip_bf16.h>

#define Bc   128
#define Sc   100
#define Wc   20
#define Dc   192
#define Lc   6
#define Hc   8
#define DFFc 256
#define DHc  24

typedef __bf16 bf16_t;
typedef __bf16 v16bf __attribute__((ext_vector_type(16)));
typedef float  v8f   __attribute__((ext_vector_type(8)));

__device__ __forceinline__ bf16_t f2bf(float f) {
    union { float f; unsigned u; } x; x.f = f;
    unsigned r = (x.u + 0x7FFFu + ((x.u >> 16) & 1u)) >> 16;
    union { unsigned short s; bf16_t b; } y; y.s = (unsigned short)r;
    return y.b;
}
__device__ __forceinline__ float bf2f(bf16_t b) {
    union { bf16_t b; unsigned short s; } x; x.b = b;
    union { unsigned u; float f; } y; y.u = ((unsigned)x.s) << 16;
    return y.f;
}

// ---------------------------------------------------------------------------
// fp32 -> bf16 conversion
// ---------------------------------------------------------------------------
__global__ void rtwp_cvt_bf16(const float* __restrict__ a, bf16_t* __restrict__ o, int n) {
    int i = blockIdx.x * blockDim.x + threadIdx.x;
    if (i < n) o[i] = f2bf(a[i]);
}

// ---------------------------------------------------------------------------
// bf16 WMMA GEMM: C[M,N] = act(A[M,K] @ Wt[N,K]^T + bias)
// One wave per 16x64 output tile: the A fragment is reused across 4 WMMAs
// per K-step (v_wmma_f32_16x16x32_bf16). M mult of 16, N mult of 64,
// K mult of 32 (true for all call sites here).
// ---------------------------------------------------------------------------
__global__ void rtwp_gemm_bf16(const bf16_t* __restrict__ A,
                               const bf16_t* __restrict__ Wt,
                               const float*  __restrict__ bias,
                               float*        __restrict__ C,
                               int M, int N, int K, int relu)
{
    int tilesN = N >> 6;
    int tiles  = (M >> 4) * tilesN;
    int wid = blockIdx.x * (blockDim.x >> 5) + (threadIdx.x >> 5);
    if (wid >= tiles) return;
    int tm = wid / tilesN, tn = wid % tilesN;
    int m0 = tm << 4,      n0 = tn << 6;

    int lane = threadIdx.x & 31;
    int half = lane >> 4;     // K-halves 0..15 / 16..31
    int l16  = lane & 15;

    const bf16_t* arow = A  + (size_t)(m0 + l16) * K + half * 16;
    const bf16_t* wrow = Wt + (size_t)(n0 + l16) * K + half * 16;
    size_t wstride = (size_t)16 * K;

    v8f acc0 = {}, acc1 = {}, acc2 = {}, acc3 = {};
    for (int k = 0; k < K; k += 32) {
        __builtin_prefetch(arow + k + 128, 0, 1);   // global_prefetch_b8
        __builtin_prefetch(wrow + k + 128, 0, 1);
        v16bf av = *(const v16bf*)(arow + k);
        v16bf b0 = *(const v16bf*)(wrow + k);
        v16bf b1 = *(const v16bf*)(wrow + wstride + k);
        v16bf b2 = *(const v16bf*)(wrow + 2 * wstride + k);
        v16bf b3 = *(const v16bf*)(wrow + 3 * wstride + k);
        acc0 = __builtin_amdgcn_wmma_f32_16x16x32_bf16(false, av, false, b0, (short)0, acc0, false, false);
        acc1 = __builtin_amdgcn_wmma_f32_16x16x32_bf16(false, av, false, b1, (short)0, acc1, false, false);
        acc2 = __builtin_amdgcn_wmma_f32_16x16x32_bf16(false, av, false, b2, (short)0, acc2, false, false);
        acc3 = __builtin_amdgcn_wmma_f32_16x16x32_bf16(false, av, false, b3, (short)0, acc3, false, false);
    }

    int rbase = m0 + half * 8;   // VGPR r: lanes0-15 -> M=r, lanes16-31 -> M=8+r
    v8f accs[4] = {acc0, acc1, acc2, acc3};
    #pragma unroll
    for (int j = 0; j < 4; ++j) {
        int col  = n0 + j * 16 + l16;
        float bb = bias ? bias[col] : 0.0f;
        #pragma unroll
        for (int r = 0; r < 8; ++r) {
            float v = accs[j][r] + bb;
            if (relu) v = fmaxf(v, 0.0f);
            C[(size_t)(rbase + r) * N + col] = v;
        }
    }
}

// ---------------------------------------------------------------------------
// Front-end feature fusion: one block per (b,s), 128 threads.
// ---------------------------------------------------------------------------
__global__ void rtwp_frontend(
    const int* __restrict__ title_idx, const int* __restrict__ class_idx,
    const int* __restrict__ process_idx, const int* __restrict__ activity_type,
    const float* __restrict__ mouse_pos, const float* __restrict__ rect,
    const float* __restrict__ flags, const float* __restrict__ kbd_in,
    const float* __restrict__ wmask,
    const float* __restrict__ title_emb, const float* __restrict__ class_emb,
    const float* __restrict__ process_emb,
    const float* __restrict__ sw1, const float* __restrict__ sb1,
    const float* __restrict__ sw2, const float* __restrict__ sb2,
    const float* __restrict__ wfw, const float* __restrict__ wfb,
    const float* __restrict__ mw1, const float* __restrict__ mb1,
    const float* __restrict__ mw2, const float* __restrict__ mb2,
    const float* __restrict__ act_emb, const float* __restrict__ kw,
    const float* __restrict__ kb, const float* __restrict__ afw,
    const float* __restrict__ afb, float* __restrict__ xout)
{
    int row = blockIdx.x;            // b*S + s
    int s   = row % Sc;
    int tid = threadIdx.x;

    __shared__ float wcomb[128];
    __shared__ float hid[64];
    __shared__ float accw[128];
    __shared__ float misc[64];
    __shared__ float afs[64];

    accw[tid] = 0.0f;
    __syncthreads();

    for (int w = 0; w < Wc; ++w) {
        int widx = row * Wc + w;
        if (tid < 32)       wcomb[tid] = title_emb[title_idx[widx] * 32 + tid];
        else if (tid < 64)  wcomb[tid] = class_emb[class_idx[widx] * 32 + (tid - 32)];
        else if (tid < 96)  wcomb[tid] = process_emb[process_idx[widx] * 32 + (tid - 64)];
        if (tid < 64) {
            float a = sb1[tid];
            #pragma unroll
            for (int j = 0; j < 4; ++j) a += sw1[tid * 6 + j] * rect[(size_t)widx * 4 + j];
            #pragma unroll
            for (int j = 0; j < 2; ++j) a += sw1[tid * 6 + 4 + j] * flags[(size_t)widx * 2 + j];
            hid[tid] = fmaxf(a, 0.0f);
        }
        __syncthreads();
        if (tid < 32) {
            float a = sb2[tid];
            for (int j = 0; j < 64; ++j) a += sw2[tid * 64 + j] * hid[j];
            wcomb[96 + tid] = a;
        }
        __syncthreads();
        float a = wfb[tid];
        for (int j = 0; j < 128; ++j) a += wfw[tid * 128 + j] * wcomb[j];
        accw[tid] += fmaxf(a, 0.0f) * wmask[widx];
        __syncthreads();
    }

    float mpx = mouse_pos[(size_t)row * 2 + 0] * (1.0f / 1920.0f);
    float mpy = mouse_pos[(size_t)row * 2 + 1] * (1.0f / 1080.0f);
    if (tid < 32)
        hid[tid] = fmaxf(mw1[tid * 2] * mpx + mw1[tid * 2 + 1] * mpy + mb1[tid], 0.0f);
    __syncthreads();
    if (tid < 32) {
        float a = mb2[tid];
        for (int j = 0; j < 32; ++j) a += mw2[tid * 32 + j] * hid[j];
        misc[tid] = a;
    } else if (tid < 48) {
        misc[tid] = act_emb[activity_type[row] * 16 + (tid - 32)];
    } else if (tid < 64) {
        misc[tid] = kw[tid - 48] * kbd_in[row] + kb[tid - 48];
    }
    __syncthreads();
    if (tid < 64) {
        float a = afb[tid];
        for (int j = 0; j < 64; ++j) a += afw[tid * 64 + j] * misc[j];
        afs[tid] = fmaxf(a, 0.0f);
    }
    __syncthreads();

    auto pe = [&](int d) {
        float fr  = __expf(-0.04797052206f * (float)(d & ~1));  // ln(1e4)/192
        float ang = (float)s * fr;
        return (d & 1) ? __cosf(ang) : __sinf(ang);
    };
    xout[(size_t)row * Dc + tid] = accw[tid] * (1.0f / Wc) + pe(tid);
    if (tid < 64)
        xout[(size_t)row * Dc + 128 + tid] = afs[tid] + pe(128 + tid);
}

// ---------------------------------------------------------------------------
// WMMA attention: one block per (b,h), 128 threads (4 waves).
// Stage q/k/v head slices into LDS via async-to-LDS gather, convert to padded
// bf16 tiles, QK^T and A*V on the WMMA pipe, softmax on 100 threads.
//   S=100 -> padded 112 rows (7 M/N tiles); DH=24 -> padded 32 (K of QK^T);
//   A*V K-dim padded to 128 (4 steps of 32); V stored transposed (32x128).
// ---------------------------------------------------------------------------
__global__ void rtwp_attn(const float* __restrict__ qkv, float* __restrict__ o)
{
    int b   = blockIdx.x / Hc;
    int h   = blockIdx.x % Hc;
    int tid = threadIdx.x;
    int wv   = tid >> 5;
    int lane = tid & 31;
    int half = lane >> 4;
    int l16  = lane & 15;

    // raw staging buffer (100 x 3 x 24 f32 = 28800B) unioned with the
    // bf16 score/probability buffer (112 x 128 bf16 = 28672B)
    __shared__ __align__(16) char smemU[28800];
    float  (*qkvs)[3][DHc] = (float(*)[3][DHc])smemU;
    bf16_t (*Sbf)[128]     = (bf16_t(*)[128])smemU;
    __shared__ __align__(16) bf16_t Qbf[112][32];
    __shared__ __align__(16) bf16_t Kbf[112][32];
    __shared__ __align__(16) bf16_t Vt[32][128];

    // ---- stage raw q/k/v slices: per-lane async gather into LDS ----------
    // 100 rows * 3 segments * 6 16B-chunks = 1800 B128 transfers
    for (int t = tid; t < Sc * 3 * 6; t += blockDim.x) {
        int si  = t / 18;
        int r   = t % 18;
        int seg = r / 6;
        int c   = r % 6;
        const float* src = qkv + (size_t)(b * Sc + si) * (3 * Dc) + seg * Dc + h * DHc + c * 4;
        unsigned long long ga = (unsigned long long)(uintptr_t)src;
        unsigned lo = (unsigned)(uintptr_t)(&qkvs[si][seg][c * 4]);
        asm volatile("global_load_async_to_lds_b128 %0, %1, off"
                     :: "v"(lo), "v"(ga) : "memory");
    }
    asm volatile("s_wait_asynccnt 0x0" ::: "memory");
    __syncthreads();

    // ---- convert to padded bf16 tiles ------------------------------------
    for (int i = tid; i < 112 * 32; i += blockDim.x) {
        int row = i >> 5, d = i & 31;
        bool ok = (row < Sc) && (d < DHc);
        Qbf[row][d] = ok ? f2bf(qkvs[row][0][d] * 0.20412414523193154f) : f2bf(0.0f);
        Kbf[row][d] = ok ? f2bf(qkvs[row][1][d]) : f2bf(0.0f);
    }
    // V transposed: Vt[d][kk]
    float vtmp[32];
    for (int i = tid; i < 32 * 128; i += blockDim.x) {
        int d = i >> 7, kk = i & 127;
        vtmp[0] = ((d < DHc) && (kk < Sc)) ? qkvs[kk][2][d] : 0.0f;
        (void)vtmp;
        // defer write: Vt aliases nothing, but qkvs aliases Sbf which is not
        // yet written, so direct write is safe here.
        Vt[d][kk] = f2bf(((d < DHc) && (kk < Sc)) ? qkvs[kk][2][d] : 0.0f);
    }
    __syncthreads();   // qkvs dead from here; smemU becomes Sbf

    // zero the score columns QK^T never writes (112..127)
    for (int i = tid; i < 112 * 16; i += blockDim.x) {
        int row = i >> 4, c = 112 + (i & 15);
        Sbf[row][c] = f2bf(0.0f);
    }
    __syncthreads();

    // ---- scores = Q @ K^T (scaled): 7x7 tiles, one WMMA each -------------
    for (int t = wv; t < 49; t += 4) {
        int m0 = (t / 7) << 4, n0 = (t % 7) << 4;
        v16bf a  = *(const v16bf*)&Qbf[m0 + l16][half * 16];
        v16bf bv = *(const v16bf*)&Kbf[n0 + l16][half * 16];
        v8f acc = {};
        acc = __builtin_amdgcn_wmma_f32_16x16x32_bf16(false, a, false, bv, (short)0, acc, false, false);
        int rbase = m0 + half * 8;
        #pragma unroll
        for (int r = 0; r < 8; ++r)
            Sbf[rbase + r][n0 + l16] = f2bf(acc[r]);
    }
    __syncthreads();

    // ---- softmax over kk<100 per row (pad K-rows give score 0 -> P 0) ----
    if (tid < Sc) {
        float mx = -1e30f;
        for (int kk = 0; kk < Sc; ++kk) mx = fmaxf(mx, bf2f(Sbf[tid][kk]));
        float sum = 0.0f;
        for (int kk = 0; kk < Sc; ++kk) {
            float p = __expf(bf2f(Sbf[tid][kk]) - mx);
            sum += p;
            Sbf[tid][kk] = f2bf(p);
        }
        float inv = 1.0f / sum;
        for (int kk = 0; kk < Sc; ++kk)
            Sbf[tid][kk] = f2bf(bf2f(Sbf[tid][kk]) * inv);
        for (int kk = Sc; kk < 112; ++kk)
            Sbf[tid][kk] = f2bf(0.0f);
    }
    __syncthreads();

    // ---- O = P @ V: 7 M-tiles x 2 N-tiles, 4 K-steps of 32 ---------------
    for (int t = wv; t < 14; t += 4) {
        int m0 = (t >> 1) << 4, n0 = (t & 1) << 4;
        v8f acc = {};
        #pragma unroll
        for (int ks = 0; ks < 128; ks += 32) {
            v16bf a  = *(const v16bf*)&Sbf[m0 + l16][ks + half * 16];
            v16bf bv = *(const v16bf*)&Vt[n0 + l16][ks + half * 16];
            acc = __builtin_amdgcn_wmma_f32_16x16x32_bf16(false, a, false, bv, (short)0, acc, false, false);
        }
        int rbase = m0 + half * 8;
        int d     = n0 + l16;
        #pragma unroll
        for (int r = 0; r < 8; ++r) {
            int row = rbase + r;
            if (row < Sc && d < DHc)
                o[(size_t)(b * Sc + row) * Dc + h * DHc + d] = acc[r];
        }
    }
}

// ---------------------------------------------------------------------------
// Residual add + LayerNorm over D=192: one block (192 threads) per row.
// ---------------------------------------------------------------------------
__global__ void rtwp_add_ln(const float* __restrict__ base, const float* __restrict__ res,
                            const float* __restrict__ g, const float* __restrict__ bta,
                            float* __restrict__ out)
{
    int row = blockIdx.x;
    int tid = threadIdx.x;
    __shared__ float rs[Dc];
    __shared__ float rq[Dc];
    size_t idx = (size_t)row * Dc + tid;
    float v = base[idx] + res[idx];
    rs[tid] = v; rq[tid] = v * v;
    __syncthreads();
    if (tid < 64) {
        rs[tid] += rs[tid + 64] + rs[tid + 128];
        rq[tid] += rq[tid + 64] + rq[tid + 128];
    }
    __syncthreads();
    for (int st = 32; st > 0; st >>= 1) {
        if (tid < st) { rs[tid] += rs[tid + st]; rq[tid] += rq[tid + st]; }
        __syncthreads();
    }
    float mean = rs[0] * (1.0f / Dc);
    float var  = fmaxf(rq[0] * (1.0f / Dc) - mean * mean, 0.0f);
    out[idx] = (v - mean) * rsqrtf(var + 1e-5f) * g[tid] + bta[tid];
}

// ---------------------------------------------------------------------------
// slot = last-token hidden + 0.1*noise, layout (W,B,D)
// ---------------------------------------------------------------------------
__global__ void rtwp_slot(const float* __restrict__ x, const float* __restrict__ noise,
                          float* __restrict__ slot)
{
    int i = blockIdx.x * blockDim.x + threadIdx.x;
    if (i >= Wc * Bc * Dc) return;
    int d = i % Dc;
    int b = (i / Dc) % Bc;
    slot[i] = x[(size_t)(b * Sc + Sc - 1) * Dc + d] + noise[i] * 0.1f;
}

// ---------------------------------------------------------------------------
// Decoder heads (tiny N): scalar kernels.
// ---------------------------------------------------------------------------
__global__ void rtwp_proj_head(const float* __restrict__ ph, const float* __restrict__ w2,
                               const float* __restrict__ b2, float* __restrict__ outp)
{
    int i = blockIdx.x * blockDim.x + threadIdx.x;
    if (i >= Wc * Bc) return;
    int w = i / Bc, b = i % Bc;
    const float* r = ph + (size_t)i * 256;
    #pragma unroll
    for (int j = 0; j < 4; ++j) {
        float a = b2[j];
        for (int k = 0; k < 256; ++k) a += r[k] * w2[j * 256 + k];
        outp[(size_t)(b * Wc + w) * 4 + j] = a;
    }
}

__global__ void rtwp_ex_head(const float* __restrict__ eh, const float* __restrict__ w2,
                             const float* __restrict__ b2, float* __restrict__ oute)
{
    int i = blockIdx.x * blockDim.x + threadIdx.x;
    if (i >= Wc * Bc) return;
    int w = i / Bc, b = i % Bc;
    const float* r = eh + (size_t)i * 128;
    float a = b2[0];
    for (int k = 0; k < 128; ++k) a += r[k] * w2[k];
    oute[b * Wc + w] = 1.0f / (1.0f + __expf(-a));
}

// ---------------------------------------------------------------------------
// Host orchestration
// ---------------------------------------------------------------------------
extern "C" void kernel_launch(void* const* d_in, const int* in_sizes, int n_in,
                              void* d_out, int out_size, void* d_ws, size_t ws_size,
                              hipStream_t stream) {
    (void)in_sizes; (void)n_in; (void)out_size; (void)ws_size;
    const int BS = Bc * Sc;

    // inputs (setup_inputs order)
    const int*   title_idx   = (const int*)d_in[0];
    const int*   class_idx   = (const int*)d_in[1];
    const int*   process_idx = (const int*)d_in[2];
    const int*   activity    = (const int*)d_in[3];
    const float* mouse_pos   = (const float*)d_in[4];
    const float* rect        = (const float*)d_in[5];
    const float* flags       = (const float*)d_in[6];
    const float* kbd         = (const float*)d_in[7];
    const float* wmask       = (const float*)d_in[8];
    const float* noise       = (const float*)d_in[9];
    const float* title_emb   = (const float*)d_in[10];
    const float* class_emb   = (const float*)d_in[11];
    const float* process_emb = (const float*)d_in[12];
    const float* sw1 = (const float*)d_in[13]; const float* sb1 = (const float*)d_in[14];
    const float* sw2 = (const float*)d_in[15]; const float* sb2 = (const float*)d_in[16];
    const float* wfw = (const float*)d_in[17]; const float* wfb = (const float*)d_in[18];
    const float* mw1 = (const float*)d_in[19]; const float* mb1 = (const float*)d_in[20];
    const float* mw2 = (const float*)d_in[21]; const float* mb2 = (const float*)d_in[22];
    const float* act_emb = (const float*)d_in[23];
    const float* kw  = (const float*)d_in[24]; const float* kb  = (const float*)d_in[25];
    const float* afw = (const float*)d_in[26]; const float* afb = (const float*)d_in[27];
    const float* qkv_w = (const float*)d_in[28]; const float* qkv_b = (const float*)d_in[29];
    const float* out_w = (const float*)d_in[30]; const float* out_b = (const float*)d_in[31];
    const float* ff1_w = (const float*)d_in[32]; const float* ff1_b = (const float*)d_in[33];
    const float* ff2_w = (const float*)d_in[34]; const float* ff2_b = (const float*)d_in[35];
    const float* ln1_g = (const float*)d_in[36]; const float* ln1_b = (const float*)d_in[37];
    const float* ln2_g = (const float*)d_in[38]; const float* ln2_b = (const float*)d_in[39];
    const float* pw1 = (const float*)d_in[40]; const float* pb1 = (const float*)d_in[41];
    const float* pw2 = (const float*)d_in[42]; const float* pb2 = (const float*)d_in[43];
    const float* xw1 = (const float*)d_in[44]; const float* xb1 = (const float*)d_in[45];
    const float* xw2 = (const float*)d_in[46]; const float* xb2 = (const float*)d_in[47];

    // workspace layout (floats, 64B aligned regions)
    float* wsf = (float*)d_ws;
    size_t off = 0;
    auto alloc = [&](size_t nf) { size_t r = off; off += (nf + 15) & ~(size_t)15; return r; };
    float* xb    = wsf + alloc((size_t)BS * Dc);
    float* qkvb  = wsf + alloc((size_t)BS * 3 * Dc);
    float* ob    = wsf + alloc((size_t)BS * Dc);
    float* ffb   = wsf + alloc((size_t)BS * DFFc);
    float* slotb = wsf + alloc((size_t)Wc * Bc * Dc);
    float* phb   = wsf + alloc((size_t)Wc * Bc * 256);
    float* ehb   = wsf + alloc((size_t)Wc * Bc * 128);
    bf16_t* hbf    = (bf16_t*)(wsf + alloc((size_t)BS * Dc / 2));
    bf16_t* ffbf   = (bf16_t*)(wsf + alloc((size_t)BS * DFFc / 2));
    bf16_t* slotbf = (bf16_t*)(wsf + alloc((size_t)Wc * Bc * Dc / 2));
    bf16_t* wqkvbf = (bf16_t*)(wsf + alloc((size_t)Lc * 3 * Dc * Dc / 2));
    bf16_t* woutbf = (bf16_t*)(wsf + alloc((size_t)Lc * Dc * Dc / 2));
    bf16_t* wf1bf  = (bf16_t*)(wsf + alloc((size_t)Lc * DFFc * Dc / 2));
    bf16_t* wf2bf  = (bf16_t*)(wsf + alloc((size_t)Lc * Dc * DFFc / 2));
    bf16_t* wp1bf  = (bf16_t*)(wsf + alloc((size_t)256 * Dc / 2));
    bf16_t* wx1bf  = (bf16_t*)(wsf + alloc((size_t)128 * Dc / 2));

    auto cvt = [&](const float* a, bf16_t* o, int n) {
        rtwp_cvt_bf16<<<(n + 255) / 256, 256, 0, stream>>>(a, o, n);
    };
    auto gemm = [&](const bf16_t* A, const bf16_t* Wt, const float* bias, float* C,
                    int M, int N, int K, int relu) {
        int tiles  = (M / 16) * (N / 64);
        int blocks = (tiles + 7) / 8;
        rtwp_gemm_bf16<<<blocks, 256, 0, stream>>>(A, Wt, bias, C, M, N, K, relu);
    };

    // weight conversion (every call; deterministic)
    cvt(qkv_w, wqkvbf, Lc * 3 * Dc * Dc);
    cvt(out_w, woutbf, Lc * Dc * Dc);
    cvt(ff1_w, wf1bf,  Lc * DFFc * Dc);
    cvt(ff2_w, wf2bf,  Lc * Dc * DFFc);
    cvt(pw1,   wp1bf,  256 * Dc);
    cvt(xw1,   wx1bf,  128 * Dc);

    // front-end
    rtwp_frontend<<<BS, 128, 0, stream>>>(
        title_idx, class_idx, process_idx, activity, mouse_pos, rect, flags, kbd,
        wmask, title_emb, class_emb, process_emb, sw1, sb1, sw2, sb2, wfw, wfb,
        mw1, mb1, mw2, mb2, act_emb, kw, kb, afw, afb, xb);

    // transformer layers
    for (int l = 0; l < Lc; ++l) {
        cvt(xb, hbf, BS * Dc);
        gemm(hbf, wqkvbf + (size_t)l * 3 * Dc * Dc, qkv_b + l * 3 * Dc, qkvb,
             BS, 3 * Dc, Dc, 0);
        rtwp_attn<<<Bc * Hc, 128, 0, stream>>>(qkvb, ob);
        cvt(ob, hbf, BS * Dc);
        gemm(hbf, woutbf + (size_t)l * Dc * Dc, out_b + l * Dc, ob, BS, Dc, Dc, 0);
        rtwp_add_ln<<<BS, Dc, 0, stream>>>(xb, ob, ln1_g + l * Dc, ln1_b + l * Dc, xb);
        cvt(xb, hbf, BS * Dc);
        gemm(hbf, wf1bf + (size_t)l * DFFc * Dc, ff1_b + l * DFFc, ffb, BS, DFFc, Dc, 1);
        cvt(ffb, ffbf, BS * DFFc);
        gemm(ffbf, wf2bf + (size_t)l * Dc * DFFc, ff2_b + l * Dc, ob, BS, Dc, DFFc, 0);
        rtwp_add_ln<<<BS, Dc, 0, stream>>>(xb, ob, ln2_g + l * Dc, ln2_b + l * Dc, xb);
    }

    // decoder
    int nslot = Wc * Bc * Dc;
    rtwp_slot<<<(nslot + 255) / 256, 256, 0, stream>>>(xb, noise, slotb);
    cvt(slotb, slotbf, nslot);
    gemm(slotbf, wp1bf, pb1, phb, Wc * Bc, 256, Dc, 1);
    gemm(slotbf, wx1bf, xb1, ehb, Wc * Bc, 128, Dc, 1);

    float* outp = (float*)d_out;
    rtwp_proj_head<<<(Wc * Bc + 255) / 256, 256, 0, stream>>>(phb, pw2, pb2, outp);
    rtwp_ex_head<<<(Wc * Bc + 255) / 256, 256, 0, stream>>>(ehb, xw2, xb2,
                                                            outp + (size_t)Bc * Wc * 4);
}
